// GraphConvNet_44822278701159
// MI455X (gfx1250) — matrix operand order repaired and checked by reading, not compiled, not run on hardware
//
#include <hip/hip_runtime.h>
#include <hip/hip_bf16.h>
#include <stdint.h>

typedef uint32_t u32;
typedef unsigned short u16;
typedef __bf16 bf16x16 __attribute__((ext_vector_type(16)));
typedef float f32x8 __attribute__((ext_vector_type(8)));

#define DEV __device__ __forceinline__

// ---------- small helpers ----------

DEV u16 f2bf(float x) {  // f32 -> bf16 round-to-nearest-even
  u32 u = __float_as_uint(x);
  u32 r = (u + 0x7FFFu + ((u >> 16) & 1u)) >> 16;
  return (u16)r;
}

DEV float gelu1(float v) {  // jax.nn.gelu default (tanh approximation)
  const float c = 0.7978845608028654f;
  float t = tanhf(c * (v + 0.044715f * v * v * v));
  return 0.5f * v * (1.0f + t);
}

DEV f32x8 gelu8(f32x8 x) {
  f32x8 r;
#pragma unroll
  for (int i = 0; i < 8; ++i) r[i] = gelu1(x[i]);
  return r;
}

// ---------- WMMA fragment loads ----------
// A (16x32 bf16), 16-bit A layout: lane l: M = l&15; h = l>>4.
// VGPR j holds K = (j>>2)*16 + h*8 + (j&3)*2 (pair). In dwords:
// kd = (j>>2)*8 + h*4 + (j&3)  -> two 4-dword runs at  h*4  and  8+h*4.
DEV bf16x16 load_a_frag(const u32* act, int strideDw, int row0, int kt, int lane) {
  const int l15 = lane & 15, h = lane >> 4;
  const u32* p = act + (row0 + l15) * strideDw + kt * 16 + h * 4;
  union { uint2 d[4]; bf16x16 v; } u;
  const uint2* q0 = (const uint2*)(p);
  const uint2* q1 = (const uint2*)(p + 8);
  u.d[0] = q0[0]; u.d[1] = q0[1];
  u.d[2] = q1[0]; u.d[3] = q1[1];
  return u.v;
}

// B (32x16 bf16), packed by k_prepack so each fragment is 8 contiguous
// dwords per lane:  addr = ((nt*KT + kt)*32 + lane)*8 + j.
DEV bf16x16 load_b_frag(const u32* Wpk, int KT, int nt, int kt, int lane) {
  const u32* p = Wpk + ((size_t)(nt * KT + kt) * 32 + (size_t)lane) * 8;
  union { uint4 d[2]; bf16x16 v; } u;
  const uint4* q = (const uint4*)p;
  u.d[0] = q[0]; u.d[1] = q[1];
  return u.v;
}

// 16 rows x 16 cols output tile: D f32 layout: VGPR j -> M = j + h*8, N = l15.
DEV f32x8 wmma_rowblk(const u32* act, int strideDw, int row0, const u32* Wpk,
                      int KT, int nt, const float* bias, int nvalid, int lane) {
  const int col = nt * 16 + (lane & 15);
  float b = (col < nvalid) ? bias[col] : 0.0f;
  f32x8 acc;
#pragma unroll
  for (int i = 0; i < 8; ++i) acc[i] = b;
  for (int kt = 0; kt < KT; ++kt) {
    bf16x16 a = load_a_frag(act, strideDw, row0, kt, lane);
    bf16x16 w = load_b_frag(Wpk, KT, nt, kt, lane);
    acc = __builtin_amdgcn_wmma_f32_16x16x32_bf16(false, a, false, w, (short)0,
                                                  acc, false, false);
  }
  return acc;
}

DEV void store_tile_bf16(u16* buf, int strideH, int row0, int nt, f32x8 x, int lane) {
  const int l15 = lane & 15, h = lane >> 4;
#pragma unroll
  for (int j = 0; j < 8; ++j)
    buf[(row0 + h * 8 + j) * strideH + nt * 16 + l15] = f2bf(x[j]);
}

// ---------- 3-layer MLP on a 64-row tile:  KPAD -> 128 (gelu) -> 128 (gelu) -> 16*NT2 ----------
template <int KT0, int NT2>
DEV void mlp3_run(const u32* sIn, int inStrideDw, u32* sMid,
                  const u32* W0p, const float* b0,
                  const u32* W1p, const float* b1,
                  const u32* W2p, const float* b2, int nb2, f32x8* accs) {
  const int lane = (int)(threadIdx.x & 31u);
  const int row0 = (int)(threadIdx.x >> 5) * 16;
  u16* midH = (u16*)sMid;
  f32x8 t[8];
#pragma unroll
  for (int nt = 0; nt < 8; ++nt)
    t[nt] = wmma_rowblk(sIn, inStrideDw, row0, W0p, KT0, nt, b0, 128, lane);
  __syncthreads();  // all waves done reading sIn (region may be reused as f32 out)
#pragma unroll
  for (int nt = 0; nt < 8; ++nt)
    store_tile_bf16(midH, 132, row0, nt, gelu8(t[nt]), lane);
  __syncthreads();
#pragma unroll
  for (int nt = 0; nt < 8; ++nt)
    t[nt] = wmma_rowblk(sMid, 66, row0, W1p, 4, nt, b1, 128, lane);
  __syncthreads();
#pragma unroll
  for (int nt = 0; nt < 8; ++nt)
    store_tile_bf16(midH, 132, row0, nt, gelu8(t[nt]), lane);
  __syncthreads();
#pragma unroll
  for (int nt = 0; nt < NT2; ++nt)
    accs[nt] = wmma_rowblk(sMid, 66, row0, W2p, 4, nt, b2, nb2, lane);
}

// ---------- weight prepack: f32 row-major [K x Nout] -> bf16 B-fragment order ----------
__global__ void k_prepack(const float* __restrict__ W, int K, int Nout, int KT,
                          int NT, u32* __restrict__ dst) {
  size_t idx = (size_t)blockIdx.x * blockDim.x + threadIdx.x;
  size_t total = (size_t)KT * NT * 256;
  if (idx >= total) return;
  int j = (int)(idx & 7);
  int l = (int)((idx >> 3) & 31);
  int f = (int)(idx >> 8);
  int kt = f % KT, nt = f / KT;
  int k = kt * 32 + (j >> 2) * 16 + (l >> 4) * 8 + (j & 3) * 2;
  int n0 = nt * 16 + (l & 15);
  float lo = (k < K && n0 < Nout) ? W[(size_t)k * Nout + n0] : 0.0f;
  float hi = ((k + 1) < K && n0 < Nout) ? W[(size_t)(k + 1) * Nout + n0] : 0.0f;
  dst[idx] = (u32)f2bf(lo) | ((u32)f2bf(hi) << 16);
}

__global__ void k_zero(float* __restrict__ p, size_t n) {
  size_t i = (size_t)blockIdx.x * blockDim.x + threadIdx.x;
  if (i < n) p[i] = 0.0f;
}

// ---------- embedder: rows x 3 -> 128 -> 128 -> 64 ----------
__global__ void k_embed(const float* __restrict__ src, int rows,
                        const u32* W0p, const float* b0, const u32* W1p,
                        const float* b1, const u32* W2p, const float* b2,
                        float* __restrict__ out) {
  constexpr int KPAD = 32, STH = 36, SDW = 18;
  __shared__ u32 sIn[64 * SDW];
  __shared__ u32 sMid[64 * 66];
  const int tid = (int)threadIdx.x;
  const int base = (int)blockIdx.x * 64;
  u16* inH = (u16*)sIn;
  for (int idx = tid; idx < 64 * KPAD; idx += 128) {
    int r = idx / KPAD, c = idx % KPAD;
    int row = base + r;
    float v = (row < rows && c < 3) ? src[(size_t)row * 3 + c] : 0.0f;
    inH[r * STH + c] = f2bf(v);
  }
  __syncthreads();
  f32x8 accs[4];
  mlp3_run<1, 4>(sIn, SDW, sMid, W0p, b0, W1p, b1, W2p, b2, 64, accs);
  const int lane = tid & 31, l15 = lane & 15, h = lane >> 4;
  const int row0 = (tid >> 5) * 16;
#pragma unroll
  for (int nt = 0; nt < 4; ++nt)
#pragma unroll
    for (int j = 0; j < 8; ++j) {
      int row = base + row0 + h * 8 + j;
      if (row < rows) out[(size_t)row * 64 + nt * 16 + l15] = accs[nt][j];
    }
}

// ---------- edge update: concat[e,n_s,n_r,g] -> MLP -> atomic segment-sum + residual LN ----------
__global__ void k_edge(float* e, const float* __restrict__ n,
                       const int* __restrict__ snd, const int* __restrict__ rcv,
                       const float* __restrict__ g, const u32* W0p,
                       const float* b0, const u32* W1p, const float* b1,
                       const u32* W2p, const float* b2,
                       const float* __restrict__ lns, const float* __restrict__ lnb,
                       float* recvb, int Ecnt) {
  constexpr int KPAD = 224, STH = 228, SDW = 114;
  __shared__ u32 sR0[64 * SDW];  // bf16 input tile, reused as f32 [64x64] output
  __shared__ u32 sMid[64 * 66];
  const int tid = (int)threadIdx.x;
  const int base = (int)blockIdx.x * 64;
  u16* inH = (u16*)sR0;
  for (int idx = tid; idx < 64 * KPAD; idx += 128) {
    int r = idx / KPAD, c = idx % KPAD;
    int i = base + r;
    float v = 0.0f;
    if (i < Ecnt) {
      if (c < 64)       v = e[(size_t)i * 64 + c];
      else if (c < 128) v = n[(size_t)snd[i] * 64 + (c - 64)];
      else if (c < 192) v = n[(size_t)rcv[i] * 64 + (c - 128)];
      else if (c < 196) v = g[c - 192];
    }
    inH[r * STH + c] = f2bf(v);
  }
  __syncthreads();
  f32x8 accs[4];
  mlp3_run<7, 4>(sR0, SDW, sMid, W0p, b0, W1p, b1, W2p, b2, 64, accs);
  __syncthreads();
  float* outF = (float*)sR0;
  {
    const int lane = tid & 31, l15 = lane & 15, h = lane >> 4;
    const int row0 = (tid >> 5) * 16;
#pragma unroll
    for (int nt = 0; nt < 4; ++nt)
#pragma unroll
      for (int j = 0; j < 8; ++j)
        outF[(row0 + h * 8 + j) * 64 + nt * 16 + l15] = accs[nt][j];
  }
  __syncthreads();
  if (tid < 64) {
    int i = base + tid;
    if (i < Ecnt) {
      const float* ne = outF + tid * 64;
      float* erow = e + (size_t)i * 64;
      float* rrow = recvb + (size_t)rcv[i] * 64;
      float mu = 0.0f;
      for (int c = 0; c < 64; ++c) {
        float nv = ne[c];
        atomicAdd(&rrow[c], nv);          // segment_sum of new_e (pre-LN)
        mu += erow[c] + nv;
      }
      mu *= (1.0f / 64.0f);
      float var = 0.0f;
      for (int c = 0; c < 64; ++c) {
        float d = erow[c] + ne[c] - mu;
        var += d * d;
      }
      var *= (1.0f / 64.0f);
      float inv = rsqrtf(var + 1e-6f);
      for (int c = 0; c < 64; ++c)
        erow[c] = (erow[c] + ne[c] - mu) * inv * lns[c] + lnb[c];
    }
  }
}

// ---------- node update: concat[n,recv,g] -> MLP -> residual LN ----------
__global__ void k_node(float* n, const float* __restrict__ recvb,
                       const float* __restrict__ g, const u32* W0p,
                       const float* b0, const u32* W1p, const float* b1,
                       const u32* W2p, const float* b2,
                       const float* __restrict__ lns, const float* __restrict__ lnb,
                       int Ncnt) {
  constexpr int KPAD = 160, STH = 164, SDW = 82;
  __shared__ u32 sR0[64 * SDW];
  __shared__ u32 sMid[64 * 66];
  const int tid = (int)threadIdx.x;
  const int base = (int)blockIdx.x * 64;
  u16* inH = (u16*)sR0;
  for (int idx = tid; idx < 64 * KPAD; idx += 128) {
    int r = idx / KPAD, c = idx % KPAD;
    int i = base + r;
    float v = 0.0f;
    if (i < Ncnt) {
      if (c < 64)       v = n[(size_t)i * 64 + c];
      else if (c < 128) v = recvb[(size_t)i * 64 + (c - 64)];
      else if (c < 132) v = g[c - 128];
    }
    inH[r * STH + c] = f2bf(v);
  }
  __syncthreads();
  f32x8 accs[4];
  mlp3_run<5, 4>(sR0, SDW, sMid, W0p, b0, W1p, b1, W2p, b2, 64, accs);
  __syncthreads();
  float* outF = (float*)sR0;
  {
    const int lane = tid & 31, l15 = lane & 15, h = lane >> 4;
    const int row0 = (tid >> 5) * 16;
#pragma unroll
    for (int nt = 0; nt < 4; ++nt)
#pragma unroll
      for (int j = 0; j < 8; ++j)
        outF[(row0 + h * 8 + j) * 64 + nt * 16 + l15] = accs[nt][j];
  }
  __syncthreads();
  if (tid < 64) {
    int i = base + tid;
    if (i < Ncnt) {
      const float* nn = outF + tid * 64;
      float* nrow = n + (size_t)i * 64;
      float mu = 0.0f;
      for (int c = 0; c < 64; ++c) mu += nrow[c] + nn[c];
      mu *= (1.0f / 64.0f);
      float var = 0.0f;
      for (int c = 0; c < 64; ++c) {
        float d = nrow[c] + nn[c] - mu;
        var += d * d;
      }
      var *= (1.0f / 64.0f);
      float inv = rsqrtf(var + 1e-6f);
      for (int c = 0; c < 64; ++c)
        nrow[c] = (nrow[c] + nn[c] - mu) * inv * lns[c] + lnb[c];
    }
  }
}

// ---------- decoder: 64 -> 128 -> 128 -> 3 ----------
__global__ void k_dec(const float* __restrict__ n, int rows, const u32* W0p,
                      const float* b0, const u32* W1p, const float* b1,
                      const u32* W2p, const float* b2, float* __restrict__ out) {
  constexpr int KPAD = 64, STH = 68, SDW = 34;
  __shared__ u32 sIn[64 * SDW];
  __shared__ u32 sMid[64 * 66];
  const int tid = (int)threadIdx.x;
  const int base = (int)blockIdx.x * 64;
  u16* inH = (u16*)sIn;
  for (int idx = tid; idx < 64 * KPAD; idx += 128) {
    int r = idx / KPAD, c = idx % KPAD;
    int row = base + r;
    float v = (row < rows) ? n[(size_t)row * 64 + c] : 0.0f;
    inH[r * STH + c] = f2bf(v);
  }
  __syncthreads();
  f32x8 accs[1];
  mlp3_run<2, 1>(sIn, SDW, sMid, W0p, b0, W1p, b1, W2p, b2, 3, accs);
  const int lane = tid & 31, l15 = lane & 15, h = lane >> 4;
  const int row0 = (tid >> 5) * 16;
  if (l15 < 3) {
#pragma unroll
    for (int j = 0; j < 8; ++j) {
      int row = base + row0 + h * 8 + j;
      if (row < rows) out[(size_t)row * 3 + l15] = accs[0][j];
    }
  }
}

// ---------- host launch ----------
extern "C" void kernel_launch(void* const* d_in, const int* in_sizes, int n_in,
                              void* d_out, int out_size, void* d_ws, size_t ws_size,
                              hipStream_t stream) {
  (void)n_in; (void)out_size; (void)ws_size;
  const float* nodes = (const float*)d_in[0];
  const float* efeat = (const float*)d_in[1];
  const float* gg    = (const float*)d_in[2];
  const float* enW0 = (const float*)d_in[3];  const float* enb0 = (const float*)d_in[4];
  const float* enW1 = (const float*)d_in[5];  const float* enb1 = (const float*)d_in[6];
  const float* enW2 = (const float*)d_in[7];  const float* enb2 = (const float*)d_in[8];
  const float* eeW0 = (const float*)d_in[9];  const float* eeb0 = (const float*)d_in[10];
  const float* eeW1 = (const float*)d_in[11]; const float* eeb1 = (const float*)d_in[12];
  const float* eeW2 = (const float*)d_in[13]; const float* eeb2 = (const float*)d_in[14];
  const float* We0 = (const float*)d_in[15];  const float* be0 = (const float*)d_in[16];
  const float* We1 = (const float*)d_in[17];  const float* be1 = (const float*)d_in[18];
  const float* We2 = (const float*)d_in[19];  const float* be2 = (const float*)d_in[20];
  const float* Wn0 = (const float*)d_in[21];  const float* bn0 = (const float*)d_in[22];
  const float* Wn1 = (const float*)d_in[23];  const float* bn1 = (const float*)d_in[24];
  const float* Wn2 = (const float*)d_in[25];  const float* bn2 = (const float*)d_in[26];
  const float* ln_s = (const float*)d_in[27]; const float* ln_b = (const float*)d_in[28];
  const float* dW0 = (const float*)d_in[29];  const float* db0 = (const float*)d_in[30];
  const float* dW1 = (const float*)d_in[31];  const float* db1 = (const float*)d_in[32];
  const float* dW2 = (const float*)d_in[33];  const float* db2 = (const float*)d_in[34];
  const int* snd = (const int*)d_in[35];
  const int* rcv = (const int*)d_in[36];

  const int N = in_sizes[0] / 3;
  const int E = in_sizes[1] / 3;

  char* ws = (char*)d_ws;
  size_t off = 0;
  auto carve = [&](size_t bytes) -> char* {
    off = (off + 255) & ~(size_t)255;
    char* p = ws + off;
    off += bytes;
    return p;
  };

  struct PK { const float* src; int K, Nout, KT, NT; u32* dst; };
  PK pk[27];
  int np = 0;
  auto addpk = [&](const float* src, int K, int Nout) -> u32* {
    int KT = (K + 31) / 32, NT = (Nout + 15) / 16;
    u32* dst = (u32*)carve((size_t)KT * NT * 256 * 4);
    pk[np++] = {src, K, Nout, KT, NT, dst};
    return dst;
  };

  u32* p_enW0 = addpk(enW0, 3, 128);
  u32* p_enW1 = addpk(enW1, 128, 128);
  u32* p_enW2 = addpk(enW2, 128, 64);
  u32* p_eeW0 = addpk(eeW0, 3, 128);
  u32* p_eeW1 = addpk(eeW1, 128, 128);
  u32* p_eeW2 = addpk(eeW2, 128, 64);
  u32* p_We0[3]; u32* p_We1[3]; u32* p_We2[3];
  u32* p_Wn0[3]; u32* p_Wn1[3]; u32* p_Wn2[3];
  for (int t = 0; t < 3; ++t) {
    p_We0[t] = addpk(We0 + (size_t)t * 196 * 128, 196, 128);
    p_We1[t] = addpk(We1 + (size_t)t * 128 * 128, 128, 128);
    p_We2[t] = addpk(We2 + (size_t)t * 128 * 64, 128, 64);
  }
  for (int t = 0; t < 3; ++t) {
    p_Wn0[t] = addpk(Wn0 + (size_t)t * 132 * 128, 132, 128);
    p_Wn1[t] = addpk(Wn1 + (size_t)t * 128 * 128, 128, 128);
    p_Wn2[t] = addpk(Wn2 + (size_t)t * 128 * 64, 128, 64);
  }
  u32* p_dW0 = addpk(dW0, 64, 128);
  u32* p_dW1 = addpk(dW1, 128, 128);
  u32* p_dW2 = addpk(dW2, 128, 3);

  float* nlat  = (float*)carve((size_t)N * 64 * sizeof(float));
  float* elat  = (float*)carve((size_t)E * 64 * sizeof(float));
  float* recvb = (float*)carve((size_t)N * 64 * sizeof(float));

  for (int i = 0; i < np; ++i) {
    size_t tot = (size_t)pk[i].KT * pk[i].NT * 256;
    k_prepack<<<dim3((unsigned)((tot + 255) / 256)), dim3(256), 0, stream>>>(
        pk[i].src, pk[i].K, pk[i].Nout, pk[i].KT, pk[i].NT, pk[i].dst);
  }

  const dim3 blk(128);
  k_embed<<<dim3((unsigned)((N + 63) / 64)), blk, 0, stream>>>(
      nodes, N, p_enW0, enb0, p_enW1, enb1, p_enW2, enb2, nlat);
  k_embed<<<dim3((unsigned)((E + 63) / 64)), blk, 0, stream>>>(
      efeat, E, p_eeW0, eeb0, p_eeW1, eeb1, p_eeW2, eeb2, elat);

  for (int t = 0; t < 3; ++t) {
    size_t rn = (size_t)N * 64;
    k_zero<<<dim3((unsigned)((rn + 255) / 256)), dim3(256), 0, stream>>>(recvb, rn);
    k_edge<<<dim3((unsigned)((E + 63) / 64)), blk, 0, stream>>>(
        elat, nlat, snd, rcv, gg, p_We0[t], be0 + t * 128, p_We1[t],
        be1 + t * 128, p_We2[t], be2 + t * 64, ln_s + t * 64, ln_b + t * 64,
        recvb, E);
    k_node<<<dim3((unsigned)((N + 63) / 64)), blk, 0, stream>>>(
        nlat, recvb, gg, p_Wn0[t], bn0 + t * 128, p_Wn1[t], bn1 + t * 128,
        p_Wn2[t], bn2 + t * 64, ln_s + t * 64, ln_b + t * 64, N);
  }

  k_dec<<<dim3((unsigned)((N + 63) / 64)), blk, 0, stream>>>(
      nlat, N, p_dW0, db0, p_dW1, db1, p_dW2, db2, (float*)d_out);
}
